// RandomBlockSelfAttention_8804682957225
// MI455X (gfx1250) — matrix operand order repaired and checked by reading, not compiled, not run on hardware
//
#include <hip/hip_runtime.h>

// ---------------- CDNA5 (gfx1250) types ----------------
typedef __attribute__((ext_vector_type(16))) __bf16 v16bf;
typedef __attribute__((ext_vector_type(8)))  float  v8f;

#define DMODEL 1024
#define NHEAD  16
#define DHEAD  64
#define BLK    256
#define BATCH  4
#define SEQ    4096
#define NBLK   16
#define N3     3072

#define QKV_STRIDE 16777216ull   // bf16 elements per q/k/v plane
#define OUT_STRIDE 16777216ull   // floats per output tensor
// workspace layout (ushort elements)
#define WS_XG  50331648ull       // gathered x, bf16 [64][256][1024]
#define WS_WIT 67108864ull       // W_in^T bf16  [3072][1024]
#define WS_WOT 70254592ull       // W_out^T bf16 [1024][1024]

__device__ __forceinline__ unsigned short f2bf(float f) {
  union { float f; unsigned u; } c; c.f = f;
  unsigned r = c.u + 0x7FFFu + ((c.u >> 16) & 1u);   // RNE
  return (unsigned short)(r >> 16);
}
__device__ __forceinline__ unsigned pack2(float a, float b) {
  return (unsigned)f2bf(a) | ((unsigned)f2bf(b) << 16);
}

union V16 { v16bf v; unsigned u[8]; unsigned short s[16]; };

__device__ __forceinline__ v16bf load_a_frag_global(const unsigned short* p0,
                                                    const unsigned short* p1) {
  V16 t;
  *(uint4*)&t.u[0] = *(const uint4*)p0;
  *(uint4*)&t.u[4] = *(const uint4*)p1;
  return t.v;
}

// ---- CDNA5 async copy: global -> LDS, 16B per lane, ASYNCcnt tracked ----
__device__ __forceinline__ unsigned lds_off(const void* p) {
  return (unsigned)(unsigned long long)p;   // low 32 bits of flat addr = LDS offset
}
__device__ __forceinline__ void async_b128(unsigned lds_addr, const unsigned short* g) {
  asm volatile("global_load_async_to_lds_b128 %0, %1, off"
               :: "v"(lds_addr), "v"(g) : "memory");
}
__device__ __forceinline__ void wait_async0() {
  asm volatile("s_wait_asynccnt 0" ::: "memory");
}

// =====================================================================
// Prep 1: gather(perm) + fp32->bf16 for x.  One block per permuted row.
// =====================================================================
__global__ void __launch_bounds__(256)
gather_convert(const float* __restrict__ x, const int* __restrict__ perm,
               unsigned short* __restrict__ xg)
{
  const int rowg = blockIdx.x;                 // b*4096 + s_perm
  const int b = rowg >> 12, s = rowg & 4095;
  const int src = perm[b * SEQ + s];
  const float4 v = ((const float4*)(x + ((size_t)b * SEQ + src) * DMODEL))[threadIdx.x];
  uint2 o; o.x = pack2(v.x, v.y); o.y = pack2(v.z, v.w);
  ((uint2*)(xg + (size_t)rowg * DMODEL))[threadIdx.x] = o;
}

// =====================================================================
// Prep 2: fp32 [K][N] -> bf16 [N][K] transpose via LDS tile.
// =====================================================================
__global__ void __launch_bounds__(256)
transpose_convert(const float* __restrict__ src, unsigned short* __restrict__ dst,
                  int K, int N)
{
  __shared__ float tile[64][65];
  const int n0 = blockIdx.x * 64, k0 = blockIdx.y * 64;
  const int tid = threadIdx.x;
  const int nn = tid & 63, kr = tid >> 6;
#pragma unroll
  for (int i = 0; i < 16; ++i)
    tile[kr * 16 + i][nn] = src[(size_t)(k0 + kr * 16 + i) * N + n0 + nn];
  __syncthreads();
  const int kl = (tid & 31) * 2, nr = tid >> 5;
#pragma unroll
  for (int i = 0; i < 8; ++i) {
    const int nl = nr * 8 + i;
    *(unsigned*)&dst[(size_t)(n0 + nl) * K + k0 + kl] = pack2(tile[kl][nl], tile[kl + 1][nl]);
  }
}

// =====================================================================
// Kernel 1: QKV GEMM, async-to-LDS double-buffered (K-step 64), WMMA bf16.
// =====================================================================
__global__ void __launch_bounds__(256)
qkv_kernel(const unsigned short* __restrict__ xg, const int* __restrict__ perm,
           const unsigned short* __restrict__ wt, const float* __restrict__ bi,
           unsigned short* __restrict__ ws, float* __restrict__ out)
{
  __shared__ int perm_s[BLK];
  __shared__ __align__(32) unsigned short a_lds[2][2][16][32][16]; // [buf][kc][rt][lane][i]
  __shared__ __align__(32) unsigned short b_lds[2][2][4][32][16];  // [buf][kc][nt][lane][i]

  const int tid = threadIdx.x, lane = tid & 31, w = tid >> 5;
  const int gl = lane >> 4, nn = lane & 15;
  const int t = blockIdx.x, bn = blockIdx.y;
  const int b = bn >> 4, n = bn & 15;
  const int h = t / 3, sel = t % 3;
  const int c0 = t * 64;

  perm_s[tid] = perm[b * SEQ + n * BLK + tid];
  __syncthreads();

  const unsigned ABUF = sizeof(a_lds[0]);       // 32 KB
  const unsigned ACH  = sizeof(a_lds[0][0]);    // 16 KB
  const unsigned BBUF = sizeof(b_lds[0]);       //  8 KB
  const unsigned BCH  = sizeof(b_lds[0][0]);    //  4 KB

  const unsigned short* asrc[2];
  unsigned adst[2];
#pragma unroll
  for (int r = 0; r < 2; ++r) {
    asrc[r] = xg + ((size_t)bn * BLK + (w * 2 + r) * 16 + nn) * DMODEL + gl * 8;
    adst[r] = lds_off(&a_lds[0][0][w * 2 + r][lane][0]);
  }
  const unsigned short* bsrc = wt + (size_t)(c0 + w * 16 + nn) * DMODEL + gl * 8;
  const unsigned bdst = lds_off(&b_lds[0][0][w][lane][0]);

  v8f acc[2][4];
#pragma unroll
  for (int r = 0; r < 2; ++r)
#pragma unroll
    for (int c = 0; c < 4; ++c)
#pragma unroll
      for (int v = 0; v < 8; ++v) acc[r][c][v] = 0.0f;

  auto issue = [&](int nb, int kk) {
#pragma unroll
    for (int kc = 0; kc < 2; ++kc) {
#pragma unroll
      for (int r = 0; r < 2; ++r) {
        const unsigned d = adst[r] + nb * ABUF + kc * ACH;
        const unsigned short* s = asrc[r] + kk + kc * 32;
        async_b128(d, s);
        async_b128(d + 16, s + 16);
      }
      if (w < 4) {
        const unsigned d = bdst + nb * BBUF + kc * BCH;
        const unsigned short* s = bsrc + kk + kc * 32;
        async_b128(d, s);
        async_b128(d + 16, s + 16);
      }
    }
  };

  issue(0, 0);
  int buf = 0;
  for (int kk = 0; kk < DMODEL; kk += 64) {
    wait_async0();
    __syncthreads();
    if (kk + 64 < DMODEL) issue(buf ^ 1, kk + 64);
#pragma unroll
    for (int kc = 0; kc < 2; ++kc) {
      v16bf fa[2], fb[4];
#pragma unroll
      for (int r = 0; r < 2; ++r) fa[r] = *(const v16bf*)&a_lds[buf][kc][w * 2 + r][lane][0];
#pragma unroll
      for (int c = 0; c < 4; ++c) fb[c] = *(const v16bf*)&b_lds[buf][kc][c][lane][0];
#pragma unroll
      for (int r = 0; r < 2; ++r)
#pragma unroll
        for (int c = 0; c < 4; ++c)
          acc[r][c] = __builtin_amdgcn_wmma_f32_16x16x32_bf16(
              false, fa[r], false, fb[c], (short)0, acc[r][c], false, false);
    }
    buf ^= 1;
  }

  // epilogue: bias, route q/k/v; k,v also scattered fp32 to d_out
  const size_t base = ((size_t)(bn * NHEAD + h) * BLK) * DHEAD;
#pragma unroll
  for (int c = 0; c < 4; ++c) {
    const int d = c * 16 + nn;
    const float bias = bi[c0 + d];
#pragma unroll
    for (int r = 0; r < 2; ++r)
#pragma unroll
      for (int v = 0; v < 8; ++v) {
        const int row = w * 32 + r * 16 + v + 8 * gl;
        const float val = acc[r][c][v] + bias;
        const size_t wsidx = base + (size_t)row * DHEAD + d;
        ws[(size_t)sel * QKV_STRIDE + wsidx] = f2bf(val);
        if (sel != 0) {
          const size_t orow = (size_t)b * SEQ + perm_s[row];
          const size_t off = (sel == 1) ? OUT_STRIDE : 2 * OUT_STRIDE;
          out[off + orow * DMODEL + (size_t)h * DHEAD + d] = val;
        }
      }
  }
}

// =====================================================================
// Kernel 2: block attention, flash-style online softmax per (bn, head).
// =====================================================================
__global__ void __launch_bounds__(256)
attn_kernel(unsigned short* __restrict__ ws)
{
  __shared__ __align__(32) unsigned short vb[8][4][32][16];
  __shared__ __align__(32) unsigned short pa[8][2][32][16];

  const int tid = threadIdx.x, lane = tid & 31, w = tid >> 5;
  const int gl = lane >> 4, nn = lane & 15;
  const int bnh = blockIdx.x;

  const unsigned short* qsrc = ws + (size_t)bnh * BLK * DHEAD;
  const unsigned short* ksrc = qsrc + QKV_STRIDE;
  const unsigned short* vsrc = qsrc + 2 * QKV_STRIDE;

  { // stage V -> B-frag layout (K=j, N=d)
    const int d0 = (tid & 31) * 2;
    const int jc = tid >> 5;
    const unsigned* v32 = (const unsigned*)vsrc;
    unsigned a[32];
#pragma unroll
    for (int jj = 0; jj < 32; ++jj) a[jj] = v32[(size_t)(jc * 32 + jj) * 32 + (d0 >> 1)];
#pragma unroll
    for (int g = 0; g < 2; ++g) {
      unsigned lo[8], hi[8];
#pragma unroll
      for (int i = 0; i < 8; ++i) {
        unsigned x0 = a[g * 16 + 2 * i], x1 = a[g * 16 + 2 * i + 1];
        lo[i] = (x0 & 0xffffu) | ((x1 & 0xffffu) << 16);
        hi[i] = (x0 >> 16) | (x1 & 0xffff0000u);
      }
      uint4* dl = (uint4*)&vb[jc][d0 >> 4][(d0 & 15) + 16 * g][0];
      dl[0] = make_uint4(lo[0], lo[1], lo[2], lo[3]);
      dl[1] = make_uint4(lo[4], lo[5], lo[6], lo[7]);
      uint4* dh = (uint4*)&vb[jc][d0 >> 4][(d0 & 15) + 1 + 16 * g][0];
      dh[0] = make_uint4(hi[0], hi[1], hi[2], hi[3]);
      dh[1] = make_uint4(hi[4], hi[5], hi[6], hi[7]);
    }
  }
  __syncthreads();

  v16bf fq[2][2];
#pragma unroll
  for (int r = 0; r < 2; ++r) {
    const int row = w * 32 + r * 16 + nn;
#pragma unroll
    for (int kt = 0; kt < 2; ++kt) {
      const unsigned short* p = qsrc + (size_t)row * DHEAD + kt * 32 + gl * 16;
      fq[r][kt] = load_a_frag_global(p, p + 8);
    }
  }

  float rmax[2][8], rsum[2][8];
  v8f o[2][4];
#pragma unroll
  for (int r = 0; r < 2; ++r)
#pragma unroll
    for (int v = 0; v < 8; ++v) { rmax[r][v] = -1e30f; rsum[r][v] = 0.0f; }
#pragma unroll
  for (int r = 0; r < 2; ++r)
#pragma unroll
    for (int c = 0; c < 4; ++c)
#pragma unroll
      for (int v = 0; v < 8; ++v) o[r][c][v] = 0.0f;

  for (int jc = 0; jc < 8; ++jc) {
    v16bf fk[2][2];
#pragma unroll
    for (int jt = 0; jt < 2; ++jt) {
      const int j = jc * 32 + jt * 16 + nn;
#pragma unroll
      for (int kt = 0; kt < 2; ++kt)
        fk[jt][kt] = *(const v16bf*)(ksrc + (size_t)j * DHEAD + kt * 32 + gl * 16);
    }
    v8f s[2][2];
#pragma unroll
    for (int r = 0; r < 2; ++r)
#pragma unroll
      for (int jt = 0; jt < 2; ++jt) {
#pragma unroll
        for (int v = 0; v < 8; ++v) s[r][jt][v] = 0.0f;
        s[r][jt] = __builtin_amdgcn_wmma_f32_16x16x32_bf16(
            false, fq[r][0], false, fk[jt][0], (short)0, s[r][jt], false, false);
        s[r][jt] = __builtin_amdgcn_wmma_f32_16x16x32_bf16(
            false, fq[r][1], false, fk[jt][1], (short)0, s[r][jt], false, false);
      }
#pragma unroll
    for (int r = 0; r < 2; ++r) {
      float cmax[8];
#pragma unroll
      for (int jt = 0; jt < 2; ++jt)
#pragma unroll
        for (int v = 0; v < 8; ++v) s[r][jt][v] *= 0.125f;
#pragma unroll
      for (int v = 0; v < 8; ++v) cmax[v] = fmaxf(s[r][0][v], s[r][1][v]);
#pragma unroll
      for (int m = 1; m < 16; m <<= 1)
#pragma unroll
        for (int v = 0; v < 8; ++v) cmax[v] = fmaxf(cmax[v], __shfl_xor(cmax[v], m, 32));
      float corr[8], psum[8];
#pragma unroll
      for (int v = 0; v < 8; ++v) {
        float nm = fmaxf(rmax[r][v], cmax[v]);
        corr[v] = __expf(rmax[r][v] - nm);
        rmax[r][v] = nm;
        psum[v] = 0.0f;
      }
#pragma unroll
      for (int jt = 0; jt < 2; ++jt)
#pragma unroll
        for (int v = 0; v < 8; ++v) {
          float p = __expf(s[r][jt][v] - rmax[r][v]);
          psum[v] += p;
          const int jl = jt * 16 + nn;
          const int gp = jl >> 4;
          const int ip = jl & 15;
          pa[w][r][(v + 8 * gl) + 16 * gp][ip] = f2bf(p);
        }
#pragma unroll
      for (int m = 1; m < 16; m <<= 1)
#pragma unroll
        for (int v = 0; v < 8; ++v) psum[v] += __shfl_xor(psum[v], m, 32);
#pragma unroll
      for (int v = 0; v < 8; ++v) rsum[r][v] = rsum[r][v] * corr[v] + psum[v];
#pragma unroll
      for (int c = 0; c < 4; ++c)
#pragma unroll
        for (int v = 0; v < 8; ++v) o[r][c][v] *= corr[v];
    }
    __syncthreads();
#pragma unroll
    for (int r = 0; r < 2; ++r) {
      v16bf fp = *(const v16bf*)&pa[w][r][lane][0];
#pragma unroll
      for (int c = 0; c < 4; ++c)
        o[r][c] = __builtin_amdgcn_wmma_f32_16x16x32_bf16(
            false, fp, false, *(const v16bf*)&vb[jc][c][lane][0],
            (short)0, o[r][c], false, false);
    }
    __syncthreads();
  }

#pragma unroll
  for (int r = 0; r < 2; ++r)
#pragma unroll
    for (int c = 0; c < 4; ++c)
#pragma unroll
      for (int v = 0; v < 8; ++v) {
        const int row = w * 32 + r * 16 + v + 8 * gl;
        const int d = c * 16 + nn;
        ws[(size_t)bnh * BLK * DHEAD + (size_t)row * DHEAD + d] =
            f2bf(o[r][c][v] / rsum[r][v]);
      }
}

// =====================================================================
// Kernel 3: output projection, async double-buffered (K-step 64) WMMA.
// =====================================================================
__global__ void __launch_bounds__(256)
outproj_kernel(const unsigned short* __restrict__ ws, const int* __restrict__ perm,
               const unsigned short* __restrict__ wt, const float* __restrict__ bo,
               float* __restrict__ out)
{
  __shared__ int perm_s[BLK];
  __shared__ __align__(32) unsigned short a_lds[2][2][16][32][16];
  __shared__ __align__(32) unsigned short b_lds[2][2][4][32][16];

  const int tid = threadIdx.x, lane = tid & 31, w = tid >> 5;
  const int gl = lane >> 4, nn = lane & 15;
  const int ct = blockIdx.x, bn = blockIdx.y;
  const int b = bn >> 4, n = bn & 15;
  const int c0 = ct * 64;

  perm_s[tid] = perm[b * SEQ + n * BLK + tid];
  __syncthreads();

  const unsigned ABUF = sizeof(a_lds[0]), ACH = sizeof(a_lds[0][0]);
  const unsigned BBUF = sizeof(b_lds[0]), BCH = sizeof(b_lds[0][0]);

  unsigned adst[2];
#pragma unroll
  for (int r = 0; r < 2; ++r) adst[r] = lds_off(&a_lds[0][0][w * 2 + r][lane][0]);
  const unsigned short* bsrc = wt + (size_t)(c0 + w * 16 + nn) * DMODEL + gl * 8;
  const unsigned bdst = lds_off(&b_lds[0][0][w][lane][0]);

  v8f acc[2][4];
#pragma unroll
  for (int r = 0; r < 2; ++r)
#pragma unroll
    for (int c = 0; c < 4; ++c)
#pragma unroll
      for (int v = 0; v < 8; ++v) acc[r][c][v] = 0.0f;

  auto issue = [&](int nb, int kk) {   // kk multiple of 64 -> one head per iteration
#pragma unroll
    for (int kc = 0; kc < 2; ++kc) {
#pragma unroll
      for (int r = 0; r < 2; ++r) {
        const unsigned short* p =
            ws + ((size_t)(bn * NHEAD + (kk >> 6)) * BLK + (w * 2 + r) * 16 + nn) * DHEAD +
            kc * 32 + gl * 8;
        const unsigned d = adst[r] + nb * ABUF + kc * ACH;
        async_b128(d, p);
        async_b128(d + 16, p + 16);
      }
      if (w < 4) {
        const unsigned d = bdst + nb * BBUF + kc * BCH;
        const unsigned short* s = bsrc + kk + kc * 32;
        async_b128(d, s);
        async_b128(d + 16, s + 16);
      }
    }
  };

  issue(0, 0);
  int buf = 0;
  for (int kk = 0; kk < DMODEL; kk += 64) {
    wait_async0();
    __syncthreads();
    if (kk + 64 < DMODEL) issue(buf ^ 1, kk + 64);
#pragma unroll
    for (int kc = 0; kc < 2; ++kc) {
      v16bf fa[2], fb[4];
#pragma unroll
      for (int r = 0; r < 2; ++r) fa[r] = *(const v16bf*)&a_lds[buf][kc][w * 2 + r][lane][0];
#pragma unroll
      for (int c = 0; c < 4; ++c) fb[c] = *(const v16bf*)&b_lds[buf][kc][c][lane][0];
#pragma unroll
      for (int r = 0; r < 2; ++r)
#pragma unroll
        for (int c = 0; c < 4; ++c)
          acc[r][c] = __builtin_amdgcn_wmma_f32_16x16x32_bf16(
              false, fa[r], false, fb[c], (short)0, acc[r][c], false, false);
    }
    buf ^= 1;
  }

#pragma unroll
  for (int c = 0; c < 4; ++c) {
    const int d = c * 16 + nn;
    const float bias = bo[c0 + d];
#pragma unroll
    for (int r = 0; r < 2; ++r)
#pragma unroll
      for (int v = 0; v < 8; ++v) {
        const int row = w * 32 + r * 16 + v + 8 * gl;
        out[((size_t)b * SEQ + perm_s[row]) * DMODEL + c0 + d] = acc[r][c][v] + bias;
      }
  }
}

extern "C" void kernel_launch(void* const* d_in, const int* in_sizes, int n_in,
                              void* d_out, int out_size, void* d_ws, size_t ws_size,
                              hipStream_t stream) {
  (void)in_sizes; (void)n_in; (void)out_size; (void)ws_size;
  const float* x    = (const float*)d_in[0];
  const int*   perm = (const int*)d_in[1];
  const float* Wi   = (const float*)d_in[2];
  const float* bi   = (const float*)d_in[3];
  const float* Wo   = (const float*)d_in[4];
  const float* bo   = (const float*)d_in[5];
  float* out = (float*)d_out;
  unsigned short* ws = (unsigned short*)d_ws;

  unsigned short* xg  = ws + WS_XG;
  unsigned short* wiT = ws + WS_WIT;
  unsigned short* woT = ws + WS_WOT;

  gather_convert   <<<dim3(16384),   256, 0, stream>>>(x, perm, xg);
  transpose_convert<<<dim3(48, 16),  256, 0, stream>>>(Wi, wiT, DMODEL, N3);
  transpose_convert<<<dim3(16, 16),  256, 0, stream>>>(Wo, woT, DMODEL, DMODEL);
  qkv_kernel       <<<dim3(48, 64),  256, 0, stream>>>(xg, perm, wiT, bi, ws, out);
  attn_kernel      <<<dim3(1024),    256, 0, stream>>>(ws);
  outproj_kernel   <<<dim3(16, 64),  256, 0, stream>>>(ws, perm, woT, bo, out);
}
